// ResidualGC_53386443489916
// MI455X (gfx1250) — compile-verified, matrix operand order checked by reading, not run on hardware
//
#include <hip/hip_runtime.h>
#include <hip/hip_bf16.h>

// ---------------------------------------------------------------------------
// ResidualGC pipeline for gfx1250 (MI455X):
//   3x GCNConv(relu) -> concat skip -> SAGEConv -> GCNConv classifier
// GEMMs use v_wmma_f32_16x16x32_bf16 (NT output tiles per wave to amortize
// the f32->bf16 A-fragment conversion); aggregations use f32 global atomics.
// ---------------------------------------------------------------------------

typedef __bf16 bf16_t;
typedef __attribute__((ext_vector_type(16))) __bf16 v16bf;
typedef __attribute__((ext_vector_type(8)))  float  v8f;

#define TPB 256  // 8 waves of 32

// ---------------- small utility kernels ----------------

__global__ void k_zero_f32(float* __restrict__ p, long total) {
    long i = (long)blockIdx.x * blockDim.x + threadIdx.x;
    if (i < total) p[i] = 0.0f;
}

// Wt[n][k] = bf16(W[k][n]); rows n in [Nout, Npad) are zero padding.
__global__ void k_transpose_w(const float* __restrict__ W, bf16_t* __restrict__ Wt,
                              int K, int Nout, int Npad) {
    long idx = (long)blockIdx.x * blockDim.x + threadIdx.x;
    long total = (long)Npad * K;
    if (idx >= total) return;
    int n = (int)(idx / K);
    int k = (int)(idx % K);
    Wt[(size_t)n * K + k] = (n < Nout) ? (bf16_t)W[(size_t)k * Nout + n] : (bf16_t)0.0f;
}

__global__ void k_degree(const int* __restrict__ dst, float* __restrict__ deg, int E) {
    int e = blockIdx.x * blockDim.x + threadIdx.x;
    if (e < E) atomicAdd(&deg[dst[e]], 1.0f);
}

// dinv = rsqrt(in_degree + 1)   (self-loop included, so always > 0)
__global__ void k_dinv(const float* __restrict__ deg, float* __restrict__ dinv, int n) {
    int i = blockIdx.x * blockDim.x + threadIdx.x;
    if (i < n) dinv[i] = rsqrtf(deg[i] + 1.0f);
}

// GCN self-loop init: out[i,c] = h[i,c] * dinv[i]^2   (full overwrite of out)
__global__ void k_gcn_self(const float* __restrict__ h, const float* __restrict__ dinv,
                           float* __restrict__ out, long total, int C) {
    long idx = (long)blockIdx.x * blockDim.x + threadIdx.x;
    if (idx >= total) return;
    long i = idx / C;
    float di = dinv[i];
    out[idx] = h[idx] * di * di;
}

// GCN edge scatter: one wave per edge; out[dst] += h[src] * dinv[src]*dinv[dst]
__global__ __launch_bounds__(TPB) void k_gcn_scatter(
    const int* __restrict__ src, const int* __restrict__ dst,
    const float* __restrict__ h, const float* __restrict__ dinv,
    float* __restrict__ out, int E, int C) {
    int w    = (int)(((long)blockIdx.x * TPB + threadIdx.x) >> 5);
    int lane = threadIdx.x & 31;
    if (w >= E) return;
    int s = src[w], d = dst[w];
    if (w + 1 < E)  // gfx1250 global_prefetch_b8 of next edge's source row
        __builtin_prefetch(h + (size_t)src[w + 1] * C, 0, 1);
    float nrm = dinv[s] * dinv[d];
    const float* hs = h + (size_t)s * C;
    float*       od = out + (size_t)d * C;
    for (int c = lane; c < C; c += 32) atomicAdd(od + c, hs[c] * nrm);
}

// SAGE edge scatter (un-normalized sum): acc[dst] += cat[src]
__global__ __launch_bounds__(TPB) void k_sage_scatter(
    const int* __restrict__ src, const int* __restrict__ dst,
    const float* __restrict__ cat, float* __restrict__ acc, int E, int C) {
    int w    = (int)(((long)blockIdx.x * TPB + threadIdx.x) >> 5);
    int lane = threadIdx.x & 31;
    if (w >= E) return;
    int s = src[w], d = dst[w];
    if (w + 1 < E)
        __builtin_prefetch(cat + (size_t)src[w + 1] * C, 0, 1);
    const float* cs = cat + (size_t)s * C;
    float*       ad = acc + (size_t)d * C;
    for (int c = lane; c < C; c += 32) atomicAdd(ad + c, cs[c]);
}

__global__ void k_sage_mean(float* __restrict__ acc, const float* __restrict__ deg,
                            long total, int C) {
    long idx = (long)blockIdx.x * blockDim.x + threadIdx.x;
    if (idx >= total) return;
    long i = idx / C;
    acc[idx] /= fmaxf(deg[i], 1.0f);
}

__global__ void k_bias_act(float* __restrict__ out, const float* __restrict__ b,
                           long total, int C, int do_relu) {
    long idx = (long)blockIdx.x * blockDim.x + threadIdx.x;
    if (idx >= total) return;
    float v = out[idx] + b[idx % C];
    if (do_relu) v = fmaxf(v, 0.0f);
    out[idx] = v;
}

// cat[i] = [ x[i] (128) | h[i] (256) ]
__global__ void k_concat(const float* __restrict__ x, const float* __restrict__ h,
                         float* __restrict__ cat, long total) {
    long idx = (long)blockIdx.x * blockDim.x + threadIdx.x;
    if (idx >= total) return;
    long i = idx / 384;
    int  c = (int)(idx % 384);
    cat[idx] = (c < 128) ? x[i * 128 + c] : h[i * 256 + (c - 128)];
}

// ---------------- WMMA bf16 GEMM, NT 16x16 output tiles per wave ----------------
// Out[M,ldc] (cols < Nout) = A[M,K](f32 -> bf16 in-register) @ Wt^T
//                            [+ Cin] [+ bias]
// Wt is bf16 [Npad,K] row-major (W transposed, zero-padded rows).
// One wave owns a 16(M) x NT*16(N) strip: per K-step the A fragment is built
// once (4 float4 loads + 8 v_cvt_pk_bf16_f32) and feeds NT back-to-back WMMAs.
// Per-lane A/B fragment layout (CDNA5 ISA 7.12.2, 16-bit 16x32):
//   lanes 0-15 : row/col = lane,     K chunks {0..7, 16..23}
//   lanes 16-31: row/col = lane-16,  K chunks {8..15, 24..31}
template <int NT>
__global__ __launch_bounds__(TPB) void k_gemm_bf16_wmma(
    const float* __restrict__ A, const bf16_t* __restrict__ Wt,
    const float* __restrict__ Cin, const float* __restrict__ bias,
    float* __restrict__ Out, int M, int K, int Nout, int ldc) {
    const int wave   = threadIdx.x >> 5;
    const int lane   = threadIdx.x & 31;
    const int mTiles = (M + 15) >> 4;
    const int mTile  = blockIdx.y * 8 + wave;
    if (mTile >= mTiles) return;  // wave-uniform: EXEC stays all-ones for WMMA

    const int nTileBase = blockIdx.x * NT;
    const int l16   = lane & 15;
    const int kBase = (lane < 16) ? 0 : 8;

    const int mRow = mTile * 16 + l16;
    const float* aRow = A + (size_t)((mRow < M) ? mRow : (M - 1)) * K;
    const bf16_t* bRow[NT];
#pragma unroll
    for (int t = 0; t < NT; ++t)
        bRow[t] = Wt + (size_t)((nTileBase + t) * 16 + l16) * K;

    v8f acc[NT] = {};

    for (int k0 = 0; k0 < K; k0 += 32) {
        // ---- A fragment: 2 chunks of 8 contiguous f32, convert to bf16 ----
        const float4* ap0 = (const float4*)(aRow + k0 + kBase);
        const float4* ap1 = (const float4*)(aRow + k0 + kBase + 16);
        float4 f0 = ap0[0], f1 = ap0[1];
        float4 f2 = ap1[0], f3 = ap1[1];
        v16bf av;
        av[0]  = (bf16_t)f0.x; av[1]  = (bf16_t)f0.y; av[2]  = (bf16_t)f0.z; av[3]  = (bf16_t)f0.w;
        av[4]  = (bf16_t)f1.x; av[5]  = (bf16_t)f1.y; av[6]  = (bf16_t)f1.z; av[7]  = (bf16_t)f1.w;
        av[8]  = (bf16_t)f2.x; av[9]  = (bf16_t)f2.y; av[10] = (bf16_t)f2.z; av[11] = (bf16_t)f2.w;
        av[12] = (bf16_t)f3.x; av[13] = (bf16_t)f3.y; av[14] = (bf16_t)f3.z; av[15] = (bf16_t)f3.w;

        // ---- NT B fragments (2 contiguous 16B bf16 loads each) + NT WMMAs ----
#pragma unroll
        for (int t = 0; t < NT; ++t) {
            union { uint4 u[2]; v16bf v; } bu;
            bu.u[0] = *(const uint4*)(bRow[t] + k0 + kBase);
            bu.u[1] = *(const uint4*)(bRow[t] + k0 + kBase + 16);
            acc[t] = __builtin_amdgcn_wmma_f32_16x16x32_bf16(
                         /*neg_a=*/false, av, /*neg_b=*/false, bu.v,
                         /*c_mod=*/(short)0, acc[t],
                         /*reuse_a=*/false, /*reuse_b=*/false);
        }
    }

    // ---- epilogue: C/D layout -> VGPR r holds row (base + r [+8 for hi lanes]) ----
    const int rowBase = mTile * 16 + ((lane < 16) ? 0 : 8);
#pragma unroll
    for (int t = 0; t < NT; ++t) {
        const int col = (nTileBase + t) * 16 + l16;
        if (col < Nout) {
#pragma unroll
            for (int r = 0; r < 8; ++r) {
                const int row = rowBase + r;
                if (row < M) {
                    float v = acc[t][r];
                    if (Cin)  v += Cin[(size_t)row * ldc + col];
                    if (bias) v += bias[col];
                    Out[(size_t)row * ldc + col] = v;
                }
            }
        }
    }
}

// ---------------------------------------------------------------------------
// host-side orchestration
// ---------------------------------------------------------------------------
static inline long cdiv(long a, long b) { return (a + b - 1) / b; }

extern "C" void kernel_launch(void* const* d_in, const int* in_sizes, int n_in,
                              void* d_out, int out_size, void* d_ws, size_t ws_size,
                              hipStream_t stream) {
    const int IN_CH = 128, HID = 256, CAT = 384, OUT_CH = 40, OUT_PAD = 48;

    const float* x   = (const float*)d_in[0];
    const int*   src = (const int*)d_in[1];
    const int    E   = in_sizes[1] / 2;
    const int*   dst = src + E;
    const int    N   = in_sizes[0] / IN_CH;

    const float* W0 = (const float*)d_in[2];  const float* b0 = (const float*)d_in[3];
    const float* W1 = (const float*)d_in[4];  const float* b1 = (const float*)d_in[5];
    const float* W2 = (const float*)d_in[6];  const float* b2 = (const float*)d_in[7];
    const float* Wl = (const float*)d_in[8];  const float* Wr = (const float*)d_in[9];
    const float* bc = (const float*)d_in[10];
    const float* Wo = (const float*)d_in[11]; const float* bo = (const float*)d_in[12];
    float* out = (float*)d_out;

    // ---- carve workspace (256B aligned regions) ----
    char* ws = (char*)d_ws;
    size_t off = 0;
    auto carve = [&](size_t bytes) -> void* {
        void* p = ws + off;
        off = (off + bytes + 255) & ~(size_t)255;
        return p;
    };
    float*  deg  = (float*)carve((size_t)N * 4);
    float*  dinv = (float*)carve((size_t)N * 4);
    float*  P    = (float*)carve((size_t)N * CAT * 4);  // ping
    float*  Q    = (float*)carve((size_t)N * CAT * 4);  // pong
    float*  R    = (float*)carve((size_t)N * HID * 4);  // SAGE output
    bf16_t* W0t  = (bf16_t*)carve((size_t)HID * IN_CH * 2);
    bf16_t* W1t  = (bf16_t*)carve((size_t)HID * HID * 2);
    bf16_t* W2t  = (bf16_t*)carve((size_t)HID * HID * 2);
    bf16_t* Wlt  = (bf16_t*)carve((size_t)HID * CAT * 2);
    bf16_t* Wrt  = (bf16_t*)carve((size_t)HID * CAT * 2);
    bf16_t* Wot  = (bf16_t*)carve((size_t)OUT_PAD * HID * 2);
    (void)ws_size; (void)n_in; (void)out_size;

    const int  mTiles     = (int)cdiv(N, 16);
    const long edgeBlocks = cdiv((long)E * 32, TPB);
    const unsigned mBlocks = (unsigned)cdiv(mTiles, 8);

    auto launch_transpose = [&](const float* W, bf16_t* Wt, int K, int Nout, int Npad) {
        long tot = (long)Npad * K;
        k_transpose_w<<<dim3((unsigned)cdiv(tot, TPB)), dim3(TPB), 0, stream>>>(W, Wt, K, Nout, Npad);
    };
    // 256-column GEMM: NT=4 -> 4 N-groups of 64 columns
    auto launch_gemm_h = [&](const float* A, const bf16_t* Wt, const float* Cin,
                             const float* bias, float* O, int K) {
        dim3 grid(4u, mBlocks);
        k_gemm_bf16_wmma<4><<<grid, dim3(TPB), 0, stream>>>(A, Wt, Cin, bias, O, N, K, HID, HID);
    };
    auto launch_gcn_agg = [&](const float* h, float* o, const float* bias, int C, int relu) {
        long tot = (long)N * C;
        k_gcn_self<<<dim3((unsigned)cdiv(tot, TPB)), dim3(TPB), 0, stream>>>(h, dinv, o, tot, C);
        k_gcn_scatter<<<dim3((unsigned)edgeBlocks), dim3(TPB), 0, stream>>>(src, dst, h, dinv, o, E, C);
        k_bias_act<<<dim3((unsigned)cdiv(tot, TPB)), dim3(TPB), 0, stream>>>(o, bias, tot, C, relu);
    };

    // ---- weight prep + degrees ----
    launch_transpose(W0, W0t, IN_CH, HID, HID);
    launch_transpose(W1, W1t, HID,   HID, HID);
    launch_transpose(W2, W2t, HID,   HID, HID);
    launch_transpose(Wl, Wlt, CAT,   HID, HID);
    launch_transpose(Wr, Wrt, CAT,   HID, HID);
    launch_transpose(Wo, Wot, HID, OUT_CH, OUT_PAD);

    k_zero_f32<<<dim3((unsigned)cdiv(N, TPB)), dim3(TPB), 0, stream>>>(deg, N);
    k_degree<<<dim3((unsigned)cdiv(E, TPB)), dim3(TPB), 0, stream>>>(dst, deg, E);
    k_dinv<<<dim3((unsigned)cdiv(N, TPB)), dim3(TPB), 0, stream>>>(deg, dinv, N);

    // ---- GCN layer 0: x @ W0 -> agg -> relu ----
    launch_gemm_h(x, W0t, nullptr, nullptr, Q, IN_CH);
    launch_gcn_agg(Q, P, b0, HID, 1);

    // ---- GCN layer 1 ----
    launch_gemm_h(P, W1t, nullptr, nullptr, Q, HID);
    launch_gcn_agg(Q, P, b1, HID, 1);

    // ---- GCN layer 2 ----
    launch_gemm_h(P, W2t, nullptr, nullptr, Q, HID);
    launch_gcn_agg(Q, P, b2, HID, 1);

    // ---- concat skip: Q = [x | h3] ----
    {
        long tot = (long)N * CAT;
        k_concat<<<dim3((unsigned)cdiv(tot, TPB)), dim3(TPB), 0, stream>>>(x, P, Q, tot);
    }

    // ---- SAGE: mean aggregate into P, then R = agg@Wl + cat@Wr + bc ----
    {
        long tot = (long)N * CAT;
        k_zero_f32<<<dim3((unsigned)cdiv(tot, TPB)), dim3(TPB), 0, stream>>>(P, tot);
        k_sage_scatter<<<dim3((unsigned)edgeBlocks), dim3(TPB), 0, stream>>>(src, dst, Q, P, E, CAT);
        k_sage_mean<<<dim3((unsigned)cdiv(tot, TPB)), dim3(TPB), 0, stream>>>(P, deg, tot, CAT);
    }
    launch_gemm_h(P, Wlt, nullptr, nullptr, R, CAT);   // R  = agg @ Wl
    launch_gemm_h(Q, Wrt, R,       bc,      R, CAT);   // R += cat @ Wr + bc

    // ---- final GCN classifier: R @ Wo -> agg -> + bo (no relu) ----
    {
        dim3 grid(1u, mBlocks);  // NT=3 -> one 48-column group (40 valid)
        k_gemm_bf16_wmma<3><<<grid, dim3(TPB), 0, stream>>>(R, Wot, nullptr, nullptr, P,
                                                            N, HID, OUT_CH, OUT_CH);
    }
    {
        long tot = (long)N * OUT_CH;
        k_gcn_self<<<dim3((unsigned)cdiv(tot, TPB)), dim3(TPB), 0, stream>>>(P, dinv, out, tot, OUT_CH);
        k_gcn_scatter<<<dim3((unsigned)edgeBlocks), dim3(TPB), 0, stream>>>(src, dst, P, dinv, out, E, OUT_CH);
        k_bias_act<<<dim3((unsigned)cdiv(tot, TPB)), dim3(TPB), 0, stream>>>(out, bo, tot, OUT_CH, 0);
    }
}
